// SelectiveNeuronBank_67800353735022
// MI455X (gfx1250) — compile-verified, matrix operand order checked by reading, not run on hardware
//
#include <hip/hip_runtime.h>

// ---------------------------------------------------------------------------
// SelectiveNeuronBank for MI455X (gfx1250), wave32 + WMMA bf16 + TDM staging.
// B=4, T=2048, D=1024, N=4096 hard-coded (reference shapes are fixed).
// ---------------------------------------------------------------------------

#define B_DIM 4
#define T_DIM 2048
#define D_DIM 1024
#define N_DIM 4096
#define BT    (B_DIM * T_DIM)          // 8192 rows

typedef __attribute__((ext_vector_type(16))) __bf16 v16bf;
typedef __attribute__((ext_vector_type(8)))  __bf16 v8bf;
typedef __attribute__((ext_vector_type(8)))  float  v8f;
typedef __attribute__((ext_vector_type(4)))  unsigned int v4u;
typedef __attribute__((ext_vector_type(4)))  unsigned int v4ui;
typedef __attribute__((ext_vector_type(8)))  int v8i_t;
typedef __attribute__((ext_vector_type(4)))  int v4i_t;

// LDS chunk geometry: rows of 32 dwords (64 bf16) + 4-dword pad = 36 dwords
// = 72 bf16 elements per row (144B; lane-to-lane bank offset 36 mod 64 ->
// conflict-free across a 16-lane group for ds_load_b128).
#define KCH   64                       // K-chunk in bf16 elements
#define LROW  72                       // padded LDS row stride in bf16 elems

// ---------------- helpers ----------------

__device__ __forceinline__ unsigned short f2bf(float f) {
    unsigned u = __builtin_bit_cast(unsigned, f);
    unsigned r = u + 0x7FFFu + ((u >> 16) & 1u);   // round-to-nearest-even
    return (unsigned short)(r >> 16);
}

__device__ __forceinline__ v16bf load_op(const unsigned short* p) {
    v8bf lo = *(const v8bf*)p;
    v8bf hi = *(const v8bf*)(p + 16);
    v16bf r;
#pragma unroll
    for (int i = 0; i < 8; ++i) { r[i] = lo[i]; r[i + 8] = hi[i]; }
    return r;
}

__device__ __forceinline__ v8f wmma_bf16(v16bf a, v16bf b, v8f c) {
    return __builtin_amdgcn_wmma_f32_16x16x32_bf16(
        /*neg_a=*/false, a, /*neg_b=*/false, b,
        /*c_mod=*/(short)0, c, /*reuse_a=*/false, /*reuse_b=*/false);
}

#if defined(__has_builtin)
#if __has_builtin(__builtin_amdgcn_tensor_load_to_lds)
#define USE_TDM 1
#endif
#endif

#ifdef USE_TDM
// Issue one TDM 2-D tile load: tile_dim0=32 dwords x tile_dim1=`rows` rows,
// with LDS padding of 4 dwords after every 32 dwords (row stride 36 dwords).
// All dims/strides in dword (data_size=2 -> 4B) units.
// This toolchain declares the 6-arg builtin:
//   (uint32x4 g0, int32x8 g1, int32x4 g2, int32x4 g3, int32x8 g4, i32 cpol)
__device__ __forceinline__ void tdm_load_2d(unsigned lds_addr,
                                            const unsigned short* gptr,
                                            unsigned tensor_dim0_dw,
                                            unsigned tensor_dim1,
                                            unsigned stride_dw,
                                            unsigned rows) {
    unsigned long long ga = (unsigned long long)(uintptr_t)gptr;
    v4ui g0;
    g0[0] = 1u;                                        // count=1 (valid user D#)
    g0[1] = lds_addr;                                  // LDS byte offset
    g0[2] = (unsigned)(ga & 0xFFFFFFFFu);              // global_addr[31:0]
    g0[3] = (unsigned)((ga >> 32) & 0x01FFFFFFu)       // global_addr[56:32]
          | (2u << 30);                                // type = 2 (image)
    v8i_t g1;
    g1[0] = (int)((2u << 16)      // data_size = 4B
                | (1u << 20)      // pad_enable
                | (4u << 22)      // pad_interval: 32 dwords
                | (3u << 25));    // pad_amount: 4 dwords
    g1[1] = (int)((tensor_dim0_dw & 0xFFFFu) << 16);   // tensor_dim0[15:0]
    g1[2] = (int)(((tensor_dim0_dw >> 16) & 0xFFFFu)
                | ((tensor_dim1 & 0xFFFFu) << 16));    // dim0 hi | dim1 lo
    g1[3] = (int)(((tensor_dim1 >> 16) & 0xFFFFu)
                | (32u << 16));                        // dim1 hi | tile_dim0=32dw
    g1[4] = (int)(rows & 0xFFFFu);                     // tile_dim1 = rows
    g1[5] = (int)stride_dw;                            // tensor_dim0_stride
    g1[6] = 0;
    g1[7] = 0;
    v4i_t z4 = {0, 0, 0, 0};
    v8i_t z8 = {0, 0, 0, 0, 0, 0, 0, 0};
    __builtin_amdgcn_tensor_load_to_lds(g0, g1, z4, z4, z8, 0);
}
#endif

// ---------------- conversion kernels ----------------

__global__ void convert_x_kernel(const float* __restrict__ x,
                                 unsigned short* __restrict__ xb, int n) {
    int i = blockIdx.x * 256 + threadIdx.x;
    if (i < n) xb[i] = f2bf(x[i]);
}

// w: [R, C] f32 row-major  ->  wt: [C, R] bf16 row-major (i.e. transposed)
__global__ void transpose_conv_kernel(const float* __restrict__ w,
                                      unsigned short* __restrict__ wt,
                                      int R, int C) {
    int i = blockIdx.x * 256 + threadIdx.x;
    int total = R * C;
    if (i < total) {
        int r = i / C;
        int c = i - r * C;
        wt[(size_t)c * R + r] = f2bf(w[i]);
    }
}

// ---------------- GEMM1: 4 fused projections + pointwise ----------------
// Block tile 64M x 64N, 8 waves in 2(M) x 4(N). Each wave: 2 M-subtiles x
// 4 weight matrices = 8 f32 accumulators. Per 64-wide K-chunk both the A
// panel (64x64) and the four B panels (64x64 each) are staged into LDS by
// the Tensor Data Mover, then each wave runs 16 WMMAs off ds_load_b128.

__global__ __launch_bounds__(256) void gemm1_fused_kernel(
    const unsigned short* __restrict__ x_bf,     // [BT, D]
    const unsigned short* __restrict__ wt_main,  // [N, D] (W^T)
    const unsigned short* __restrict__ wt_gate,
    const unsigned short* __restrict__ wt_mod,
    const unsigned short* __restrict__ wt_delta,
    const float* __restrict__ b_delta,           // [N]
    const float* __restrict__ A_log,             // [N]
    const float* __restrict__ sw_logit,          // [N]
    float* __restrict__ decay_out,               // [BT, N]
    float* __restrict__ update_out,              // [BT, N]
    float* __restrict__ gterm_out,               // [BT, N]
    float* __restrict__ smod_out)                // [BT, N]
{
    __shared__ unsigned short lds_a[64 * LROW];          //  9.2 KB
    __shared__ unsigned short lds_b[4][64 * LROW];       // 36.9 KB

    const int nbn     = N_DIM / 64;                      // 64 N-blocks
    const int mtile   = blockIdx.x / nbn;
    const int nblk    = blockIdx.x % nbn;
    const int tid     = threadIdx.x;
    const int wave    = tid >> 5;
    const int lane    = tid & 31;
    const int mgrp    = wave >> 2;                       // 0..1
    const int ngrp    = wave & 3;                        // 0..3

    const int rowbase = mtile * 64;
    const int nbase   = nblk * 64;

    const int arow = lane & 15;
    const int koff = (lane >> 4) << 3;                   // 0 or 8

    const unsigned short* wt[4] = {wt_main, wt_gate, wt_mod, wt_delta};

    v8f acc[4][2];
#pragma unroll
    for (int w = 0; w < 4; ++w)
#pragma unroll
        for (int m = 0; m < 2; ++m) acc[w][m] = (v8f){};

    for (int kc = 0; kc < D_DIM; kc += KCH) {
        __syncthreads();   // previous chunk fully consumed before overwrite
#ifdef USE_TDM
        if (wave == 0) {
            tdm_load_2d((unsigned)(uintptr_t)(void*)lds_a,
                        x_bf + (size_t)rowbase * D_DIM + kc,
                        D_DIM / 2, BT, D_DIM / 2, 64);
#pragma unroll
            for (int w = 0; w < 4; ++w)
                tdm_load_2d((unsigned)(uintptr_t)(void*)lds_b[w],
                            wt[w] + (size_t)nbase * D_DIM + kc,
                            D_DIM / 2, N_DIM, D_DIM / 2, 64);
            __builtin_amdgcn_s_wait_tensorcnt(0);
        }
#else
        // Fallback: cooperative vector staging into the same padded layout.
        for (int c = tid; c < 2560; c += 256) {
            unsigned short* dst;
            const unsigned short* src;
            if (c < 512) {
                int row = c >> 3, kk = (c & 7) << 3;
                dst = &lds_a[row * LROW + kk];
                src = &x_bf[(size_t)(rowbase + row) * D_DIM + kc + kk];
            } else {
                int cc = c - 512;
                int w = cc >> 9, row = (cc >> 3) & 63, kk = (cc & 7) << 3;
                dst = &lds_b[w][row * LROW + kk];
                src = &wt[w][(size_t)(nbase + row) * D_DIM + kc + kk];
            }
            *(v4u*)dst = *(const v4u*)src;
        }
#endif
        __syncthreads();

#pragma unroll
        for (int ks = 0; ks < 2; ++ks) {
            const int kb = ks * 32 + koff;
            v16bf a0 = load_op(&lds_a[(mgrp * 32 + arow) * LROW + kb]);
            v16bf a1 = load_op(&lds_a[(mgrp * 32 + 16 + arow) * LROW + kb]);
#pragma unroll
            for (int w = 0; w < 4; ++w) {
                v16bf b = load_op(&lds_b[w][(ngrp * 16 + arow) * LROW + kb]);
                acc[w][0] = wmma_bf16(a0, b, acc[w][0]);
                acc[w][1] = wmma_bf16(a1, b, acc[w][1]);
            }
        }
    }

    // Pointwise epilogue. C/D layout: VGPR r -> M = r + 8*(lane>=16), N = lane&15.
    const int ncol = nbase + ngrp * 16 + (lane & 15);
    const float bdel = b_delta[ncol];
    const float An   = -__expf(A_log[ncol]);
    const float sw   = 1.0f / (1.0f + __expf(-sw_logit[ncol]));
    const int   mofs = (lane >> 4) << 3;

#pragma unroll
    for (int m = 0; m < 2; ++m) {
#pragma unroll
        for (int r = 0; r < 8; ++r) {
            int rowg = rowbase + mgrp * 32 + m * 16 + r + mofs;
            size_t idx = (size_t)rowg * N_DIM + ncol;

            float raw = acc[0][m][r];
            float dl  = acc[3][m][r] + bdel;
            float delta = (dl > 20.0f) ? dl : log1pf(__expf(dl));
            float dec   = __expf(delta * An);
            float gate  = 1.0f / (1.0f + __expf(-acc[1][m][r]));
            float upd   = (1.0f - dec) * gate * tanhf(raw);
            float gelu  = 0.5f * raw * (1.0f + erff(raw * 0.70710678f));
            float onep  = 1.0f + tanhf(acc[2][m][r]);

            decay_out[idx]  = dec;
            update_out[idx] = upd;
            gterm_out[idx]  = (1.0f - sw) * gelu * onep;  // (1-sw)*gelu(raw)*(1+mod)
            smod_out[idx]   = sw * onep;                  // sw*(1+mod)
        }
    }
}

// ---------------- scan: linear recurrence along T ----------------

__global__ __launch_bounds__(256) void scan_kernel(
    const float* __restrict__ state0,   // [B, N]
    const float* __restrict__ decay,    // [BT, N]
    const float* __restrict__ update,
    const float* __restrict__ gterm,
    const float* __restrict__ smod,
    unsigned short* __restrict__ y_bf,  // [BT, N] (bf16, GEMM2 input)
    float* __restrict__ final_state)    // [B, N]
{
    int tid = blockIdx.x * 256 + threadIdx.x;     // 0 .. B*N-1
    int b = tid / N_DIM;
    int n = tid - b * N_DIM;

    float s = state0[tid];
    size_t base = (size_t)b * T_DIM * N_DIM + n;

    for (int t = 0; t < T_DIM; ++t) {
        size_t idx = base + (size_t)t * N_DIM;
        if (t + 16 < T_DIM) {
            size_t pidx = idx + (size_t)16 * N_DIM;
            __builtin_prefetch(&decay[pidx], 0, 1);
            __builtin_prefetch(&update[pidx], 0, 1);
            __builtin_prefetch(&gterm[pidx], 0, 1);
            __builtin_prefetch(&smod[pidx], 0, 1);
        }
        s = decay[idx] * s + update[idx];
        float y = gterm[idx] + smod[idx] * s;
        y_bf[idx] = f2bf(y);
    }
    final_state[tid] = s;
}

// ---------------- GEMM2: output projection ----------------
// out[BT, D] = y[BT, N] @ W_out[N, D], using Wout^T [D, N] bf16.
// Block tile 64M x 128D, 8 waves in 2(M) x 4(D); each wave 2x2 accumulators.
// A (64xKCH) and B (128xKCH) panels TDM-staged into LDS per K-chunk.

__global__ __launch_bounds__(256) void gemm2_kernel(
    const unsigned short* __restrict__ y_bf,   // [BT, N]
    const unsigned short* __restrict__ woutt,  // [D, N] (W_out^T)
    float* __restrict__ out)                   // [BT, D]
{
    __shared__ unsigned short lds_a[64 * LROW];          //  9.2 KB
    __shared__ unsigned short lds_b[128 * LROW];         // 18.4 KB

    const int nbd   = D_DIM / 128;             // 8 D-blocks
    const int mtile = blockIdx.x / nbd;
    const int dblk  = blockIdx.x % nbd;
    const int tid   = threadIdx.x;
    const int wave  = tid >> 5;
    const int lane  = tid & 31;
    const int mgrp  = wave >> 2;               // 0..1
    const int ngrp  = wave & 3;                // 0..3

    const int rowbase = mtile * 64;
    const int dbase   = dblk * 128;
    const int arow = lane & 15;
    const int koff = (lane >> 4) << 3;

    v8f acc[2][2];
#pragma unroll
    for (int ms = 0; ms < 2; ++ms)
#pragma unroll
        for (int ns = 0; ns < 2; ++ns) acc[ms][ns] = (v8f){};

    for (int kc = 0; kc < N_DIM; kc += KCH) {
        __syncthreads();
#ifdef USE_TDM
        if (wave == 0) {
            tdm_load_2d((unsigned)(uintptr_t)(void*)lds_a,
                        y_bf + (size_t)rowbase * N_DIM + kc,
                        N_DIM / 2, BT, N_DIM / 2, 64);
            tdm_load_2d((unsigned)(uintptr_t)(void*)lds_b,
                        woutt + (size_t)dbase * N_DIM + kc,
                        N_DIM / 2, D_DIM, N_DIM / 2, 128);
            __builtin_amdgcn_s_wait_tensorcnt(0);
        }
#else
        for (int c = tid; c < 1536; c += 256) {
            unsigned short* dst;
            const unsigned short* src;
            if (c < 512) {
                int row = c >> 3, kk = (c & 7) << 3;
                dst = &lds_a[row * LROW + kk];
                src = &y_bf[(size_t)(rowbase + row) * N_DIM + kc + kk];
            } else {
                int cc = c - 512;
                int row = cc >> 3, kk = (cc & 7) << 3;
                dst = &lds_b[row * LROW + kk];
                src = &woutt[(size_t)(dbase + row) * N_DIM + kc + kk];
            }
            *(v4u*)dst = *(const v4u*)src;
        }
#endif
        __syncthreads();

#pragma unroll
        for (int ks = 0; ks < 2; ++ks) {
            const int kb = ks * 32 + koff;
            v16bf a0 = load_op(&lds_a[(mgrp * 32 + arow) * LROW + kb]);
            v16bf a1 = load_op(&lds_a[(mgrp * 32 + 16 + arow) * LROW + kb]);
            v16bf b0 = load_op(&lds_b[(ngrp * 32 + arow) * LROW + kb]);
            v16bf b1 = load_op(&lds_b[(ngrp * 32 + 16 + arow) * LROW + kb]);
            acc[0][0] = wmma_bf16(a0, b0, acc[0][0]);
            acc[1][0] = wmma_bf16(a1, b0, acc[1][0]);
            acc[0][1] = wmma_bf16(a0, b1, acc[0][1]);
            acc[1][1] = wmma_bf16(a1, b1, acc[1][1]);
        }
    }

    const int mofs = (lane >> 4) << 3;
#pragma unroll
    for (int ms = 0; ms < 2; ++ms)
#pragma unroll
        for (int ns = 0; ns < 2; ++ns) {
            int dcol = dbase + ngrp * 32 + ns * 16 + (lane & 15);
#pragma unroll
            for (int r = 0; r < 8; ++r) {
                int rowg = rowbase + mgrp * 32 + ms * 16 + r + mofs;
                out[(size_t)rowg * D_DIM + dcol] = acc[ms][ns][r];
            }
        }
}

// ---------------- launcher ----------------

extern "C" void kernel_launch(void* const* d_in, const int* in_sizes, int n_in,
                              void* d_out, int out_size, void* d_ws, size_t ws_size,
                              hipStream_t stream) {
    (void)in_sizes; (void)n_in; (void)out_size; (void)ws_size;

    const float* x        = (const float*)d_in[0];   // [B,T,D]
    const float* state0   = (const float*)d_in[1];   // [B,N]
    const float* W_main   = (const float*)d_in[2];   // [D,N]
    const float* W_gate   = (const float*)d_in[3];
    const float* W_mod    = (const float*)d_in[4];
    const float* W_delta  = (const float*)d_in[5];
    const float* b_delta  = (const float*)d_in[6];   // [N]
    const float* W_out    = (const float*)d_in[7];   // [N,D]
    const float* A_log    = (const float*)d_in[8];   // [N]
    const float* sw_logit = (const float*)d_in[9];   // [N]

    float* out_main  = (float*)d_out;                          // [B,T,D]
    float* out_state = (float*)d_out + (size_t)BT * D_DIM;     // [B,N]

    // Workspace carve-up (byte offsets, all >=256B aligned).
    char* ws = (char*)d_ws;
    size_t off = 0;
    auto take = [&](size_t bytes) { char* p = ws + off; off += (bytes + 255) & ~(size_t)255; return p; };

    unsigned short* x_bf    = (unsigned short*)take((size_t)BT * D_DIM * 2);      // 16 MB
    unsigned short* wt_main = (unsigned short*)take((size_t)N_DIM * D_DIM * 2);   // 8 MB
    unsigned short* wt_gate = (unsigned short*)take((size_t)N_DIM * D_DIM * 2);
    unsigned short* wt_mod  = (unsigned short*)take((size_t)N_DIM * D_DIM * 2);
    unsigned short* wt_del  = (unsigned short*)take((size_t)N_DIM * D_DIM * 2);
    unsigned short* wout_t  = (unsigned short*)take((size_t)D_DIM * N_DIM * 2);   // 8 MB
    float* decay  = (float*)take((size_t)BT * N_DIM * 4);                         // 128 MB
    float* update = (float*)take((size_t)BT * N_DIM * 4);
    float* gterm  = (float*)take((size_t)BT * N_DIM * 4);
    float* smod   = (float*)take((size_t)BT * N_DIM * 4);
    unsigned short* y_bf = (unsigned short*)take((size_t)BT * N_DIM * 2);         // 64 MB

    // 1) precision conversion + weight transposes (one-shot, ~176MB traffic)
    {
        int n = BT * D_DIM;                                   // 8.4M
        convert_x_kernel<<<(n + 255) / 256, 256, 0, stream>>>(x, x_bf, n);
        int nw = D_DIM * N_DIM;                               // 4.2M
        int g = (nw + 255) / 256;
        transpose_conv_kernel<<<g, 256, 0, stream>>>(W_main,  wt_main, D_DIM, N_DIM);
        transpose_conv_kernel<<<g, 256, 0, stream>>>(W_gate,  wt_gate, D_DIM, N_DIM);
        transpose_conv_kernel<<<g, 256, 0, stream>>>(W_mod,   wt_mod,  D_DIM, N_DIM);
        transpose_conv_kernel<<<g, 256, 0, stream>>>(W_delta, wt_del,  D_DIM, N_DIM);
        transpose_conv_kernel<<<g, 256, 0, stream>>>(W_out,   wout_t,  N_DIM, D_DIM);
    }

    // 2) fused 4-way projection GEMM + pointwise (TDM-staged, 64x64 blocks)
    {
        dim3 grid((BT / 64) * (N_DIM / 64));                  // 128 * 64 = 8192
        gemm1_fused_kernel<<<grid, 256, 0, stream>>>(
            x_bf, wt_main, wt_gate, wt_mod, wt_del,
            b_delta, A_log, sw_logit,
            decay, update, gterm, smod);
    }

    // 3) linear recurrence scan along T; emits bf16 GEMM2 input + final state
    {
        dim3 grid((B_DIM * N_DIM) / 256);                     // 64 blocks
        scan_kernel<<<grid, 256, 0, stream>>>(
            state0, decay, update, gterm, smod, y_bf, out_state);
    }

    // 4) output projection GEMM (TDM-staged, 64x128 blocks)
    {
        dim3 grid((BT / 64) * (D_DIM / 128));                 // 128 * 8 = 1024
        gemm2_kernel<<<grid, 256, 0, stream>>>(y_bf, wout_t, out_main);
    }
}
